// EdgePointGNN_50225347560184
// MI455X (gfx1250) — matrix operand order repaired and checked by reading, not compiled, not run on hardware
//
#include <hip/hip_runtime.h>
#include <hip/hip_bf16.h>
#include <math.h>

typedef __attribute__((ext_vector_type(16))) _Float16 v16h;
typedef __attribute__((ext_vector_type(8)))  float    v8f;

#define NN 20000
#define NE 160000

// ---------------- fragment index helpers (CDNA5 WMMA f16 layouts) -------------
// A-matrix 16x32 (MxK) 16-bit: lane = m + 16*h ; VGPR p holds K pair
//   k0 = 2p (+8 if p>=4) + 8h   (ISA 7.12.2 table)
// Half index: kt*512 + lane*16 + p*2 + pair
__device__ __forceinline__ int afrag_idx(int m, int k) {
  int kt = k >> 5, kk = k & 31;
  int h  = (kk >> 3) & 1;
  int p  = ((kk & 7) >> 1) + ((kk & 16) ? 4 : 0);
  return kt * 512 + (m + 16 * h) * 16 + p * 2 + (kk & 1);
}
// Dword index for an even column c: columns (c, c+1) share one dword.
__device__ __forceinline__ int afrag_dw(int m, int c) {
  int kt = c >> 5, kk = c & 31;
  int h  = (kk >> 3) & 1;
  int p  = ((kk & 7) >> 1) + ((kk & 16) ? 4 : 0);
  return kt * 256 + (m + 16 * h) * 8 + p;
}
__device__ __forceinline__ unsigned int pack_h2(float a, float b) {
  union { _Float16 f[2]; unsigned int u; } cv;
  cv.f[0] = (_Float16)a; cv.f[1] = (_Float16)b;
  return cv.u;
}

// ---------------- per-wave GEMM: D[16 x NT*16] += A[16 x KT*32] * W ----------
template<int KT, int NT>
__device__ __forceinline__ void wave_gemm(const _Float16* A, const unsigned int* W,
                                          v8f acc[NT], int lane) {
  for (int kt = 0; kt < KT; ++kt) {
    v16h a = *(const v16h*)(A + (kt * 32 + lane) * 16);
#pragma unroll
    for (int t = 0; t < NT; ++t) {
      v16h b = *(const v16h*)((const _Float16*)W + ((kt * NT + t) * 32 + lane) * 16);
      acc[t] = __builtin_amdgcn_wmma_f32_16x16x32_f16(false, a, false, b,
                                                      (short)0, acc[t], false, false);
    }
  }
}

// bias + LayerNorm(width NT*16) + ReLU on D-layout accumulators, write next
// A-fragment (f16) into LDS.
template<int NT>
__device__ __forceinline__ void ln_relu_store(v8f acc[NT], const float* __restrict__ bias,
                                              const float* __restrict__ g,
                                              const float* __restrict__ be,
                                              _Float16* outA, int lane) {
  const float invW = 1.0f / (float)(NT * 16);
  const int nl = lane & 15, h = (lane >> 4) & 1;
  float bv[NT], gv[NT], bev[NT];
#pragma unroll
  for (int t = 0; t < NT; ++t) { bv[t] = bias[t*16+nl]; gv[t] = g[t*16+nl]; bev[t] = be[t*16+nl]; }
#pragma unroll
  for (int r = 0; r < 8; ++r) {
    float vals[NT], s = 0.f, s2 = 0.f;
#pragma unroll
    for (int t = 0; t < NT; ++t) { float v = acc[t][r] + bv[t]; vals[t] = v; s += v; s2 += v*v; }
#pragma unroll
    for (int o = 1; o < 16; o <<= 1) { s += __shfl_xor(s, o, 32); s2 += __shfl_xor(s2, o, 32); }
    float mean = s * invW;
    float var  = s2 * invW - mean * mean;
    float inv  = rsqrtf(var + 1e-5f);
    int m = r + 8 * h;
#pragma unroll
    for (int t = 0; t < NT; ++t) {
      float v = (vals[t] - mean) * inv * gv[t] + bev[t];
      v = fmaxf(v, 0.f);
      outA[afrag_idx(m, t * 16 + nl)] = (_Float16)v;
    }
  }
}

// ---------------- parameter bundles ------------------------------------------
struct MlpPtrs {
  const unsigned int *w1, *w2, *w3;
  const float *b1, *g1, *be1, *b2, *g2, *be2, *b3;
};
struct EdgeParams { MlpPtrs m[4]; };
struct NodeParams { MlpPtrs fc, gh; };
struct MlpHost { const float *W1,*W2,*W3,*b1,*g1,*be1,*b2,*g2,*be2,*b3; };

// ---------------- edge MLP layer (4 unshared MLPs, one 16-edge tile / wave) --
template<int F, int KT1>
__global__ __launch_bounds__(64) void edge_mlp_kernel(const float* __restrict__ xs,
    const int* __restrict__ src, const int* __restrict__ dst, EdgeParams P,
    float* __restrict__ Ssum, float* __restrict__ Smax) {
  __shared__ __attribute__((aligned(32))) unsigned int aT[2][KT1 * 256];
  __shared__ __attribute__((aligned(32))) unsigned int hT[2][2 * 256];
  const int wave = threadIdx.x >> 5, lane = threadIdx.x & 31;
  const int e0 = (blockIdx.x * 2 + wave) * 16;
  unsigned int* A  = aT[wave];
  unsigned int* Hd = hT[wave];
  const int DIN = 2 * F - 2;
  // ---- build A tile: m_in = [x_i (F), x_j[3:] (F-3), dist2 (1), 0 pad] ----
  // Each lane writes packed (c, c+1) column pairs with one ds_store_b32.
  for (int e = 0; e < 16; ++e) {
    const int eid = e0 + e;
    const int ni = dst[eid], nj = src[eid];
    const float* xi = xs + (size_t)ni * F;
    const float* xj = xs + (size_t)nj * F;
    float d = 0.f;
    if (lane < 3) { float t = xj[lane] - xi[lane]; d = t * t; }
    d += __shfl_xor(d, 1, 32);
    d += __shfl_xor(d, 2, 32);
    const float dist2 = __shfl(d, 0, 32);
    for (int c = lane * 2; c < KT1 * 32; c += 64) {
      float v0, v1;
      if (c < F)             v0 = xi[c];
      else if (c < DIN - 1)  v0 = xj[c - F + 3];
      else if (c == DIN - 1) v0 = dist2;
      else                   v0 = 0.f;
      const int c1 = c + 1;
      if (c1 < F)             v1 = xi[c1];
      else if (c1 < DIN - 1)  v1 = xj[c1 - F + 3];
      else if (c1 == DIN - 1) v1 = dist2;
      else                    v1 = 0.f;
      A[afrag_dw(e, c)] = pack_h2(v0, v1);
    }
  }
  __syncthreads();
  const int nl = lane & 15, hh = lane >> 4;
  // hoist destination node indices for the scatter phase (8 rows per lane)
  int nd[8];
#pragma unroll
  for (int r = 0; r < 8; ++r) nd[r] = dst[e0 + r + 8 * hh];
#pragma unroll
  for (int u = 0; u < 4; ++u) {
    const MlpPtrs& mp = P.m[u];
    v8f acc1[4] = {};
    wave_gemm<KT1, 4>((const _Float16*)A, mp.w1, acc1, lane);
    ln_relu_store<4>(acc1, mp.b1, mp.g1, mp.be1, (_Float16*)Hd, lane);
    __syncthreads();
    v8f acc2[4] = {};
    wave_gemm<2, 4>((const _Float16*)Hd, mp.w2, acc2, lane);
    __syncthreads();
    ln_relu_store<4>(acc2, mp.b2, mp.g2, mp.be2, (_Float16*)Hd, lane);
    __syncthreads();
    v8f acc3[2] = {};
    wave_gemm<2, 2>((const _Float16*)Hd, mp.w3, acc3, lane);
    // bias + scatter-reduce (sum & max) by dst node
#pragma unroll
    for (int t = 0; t < 2; ++t) {
      const int c = t * 16 + nl;
      const float b3 = mp.b3[c];
#pragma unroll
      for (int r = 0; r < 8; ++r) {
        const float v = acc3[t][r] + b3;
        const int node = nd[r];
        atomicAdd(&Ssum[(size_t)node * 128 + u * 32 + c], v);
        float* pm = &Smax[(size_t)node * 128 + u * 32 + c];
        if (v >= 0.f) atomicMax((int*)pm, __float_as_int(v));
        else          atomicMin((unsigned int*)pm, __float_as_uint(v));
      }
    }
    __syncthreads();
  }
}

// ---------------- final node heads: out = mlp3(xr, fc) + mlp3(x, gh) ---------
__global__ __launch_bounds__(64) void node_out_kernel(const float* __restrict__ xr,
    const float* __restrict__ x, NodeParams P, float* __restrict__ out) {
  __shared__ __attribute__((aligned(32))) unsigned int aF[2][12 * 256];
  __shared__ __attribute__((aligned(32))) unsigned int aG[2][256];
  __shared__ __attribute__((aligned(32))) unsigned int hT[2][256];
  const int wave = threadIdx.x >> 5, lane = threadIdx.x & 31;
  const int n0 = (blockIdx.x * 2 + wave) * 16;
  unsigned int *AF = aF[wave], *AG = aG[wave], *Hd = hT[wave];
  for (int e = 0; e < 16; ++e) {
    const float* rf = xr + (size_t)(n0 + e) * 384;
    for (int c = lane * 2; c < 384; c += 64)
      AF[afrag_dw(e, c)] = pack_h2(rf[c], rf[c + 1]);
    const float* rg = x + (size_t)(n0 + e) * 7;
    if (lane < 16) {
      const int c = lane * 2;
      float v0 = (c < 7)     ? rg[c]     : 0.f;
      float v1 = (c + 1 < 7) ? rg[c + 1] : 0.f;
      AG[afrag_dw(e, c)] = pack_h2(v0, v1);
    }
  }
  __syncthreads();
  const int nl = lane & 15, hh = lane >> 4;
  float res[8];
#pragma unroll
  for (int r = 0; r < 8; ++r) res[r] = 0.f;
  for (int head = 0; head < 2; ++head) {
    const MlpPtrs& p = head ? P.gh : P.fc;
    v8f a1[2] = {};
    if (head) wave_gemm<1, 2>((const _Float16*)AG, p.w1, a1, lane);
    else      wave_gemm<12, 2>((const _Float16*)AF, p.w1, a1, lane);
    ln_relu_store<2>(a1, p.b1, p.g1, p.be1, (_Float16*)Hd, lane);
    __syncthreads();
    v8f a2[2] = {};
    wave_gemm<1, 2>((const _Float16*)Hd, p.w2, a2, lane);
    __syncthreads();
    ln_relu_store<2>(a2, p.b2, p.g2, p.be2, (_Float16*)Hd, lane);
    __syncthreads();
    v8f a3[1] = {};
    wave_gemm<1, 1>((const _Float16*)Hd, p.w3, a3, lane);
    const float b3 = (nl < 4) ? p.b3[nl] : 0.f;
#pragma unroll
    for (int r = 0; r < 8; ++r) res[r] += a3[0][r] + b3;
    __syncthreads();
  }
  if (nl < 4) {
#pragma unroll
    for (int r = 0; r < 8; ++r) {
      const int m = r + 8 * hh;
      out[(size_t)(n0 + m) * 4 + nl] = res[r];
    }
  }
}

// ---------------- small utility kernels --------------------------------------
// Pack fp32 W[KxN] (row-major) into f16 B-fragment order, zero-padded.
// B layout (16-bit, 32xK tile): lane n = lane&15 ; half h,
// VGPR p holds K = kt*32 + 8h + 2p (+8 if p>=4) (+1).
__global__ void pack_w_kernel(const float* __restrict__ W, unsigned int* __restrict__ dst,
                              int K, int N, int KT, int NT) {
  int idx = blockIdx.x * blockDim.x + threadIdx.x;
  int total = KT * NT * 256;
  if (idx >= total) return;
  int p = idx & 7, lane = (idx >> 3) & 31, tn = idx >> 8;
  int t = tn % NT, kt = tn / NT;
  int h = lane >> 4, n = t * 16 + (lane & 15);
  int k0 = kt * 32 + 8 * h + 2 * p + (p >= 4 ? 8 : 0);
  float lo = (k0     < K && n < N) ? W[(size_t)k0 * N + n]       : 0.f;
  float hi = (k0 + 1 < K && n < N) ? W[(size_t)(k0 + 1) * N + n] : 0.f;
  dst[idx] = pack_h2(lo, hi);
}

__global__ void fill_f32(float* p, float v, int n) {
  int i = blockIdx.x * blockDim.x + threadIdx.x; if (i < n) p[i] = v;
}
__global__ void fill_i32(int* p, int v, int n) {
  int i = blockIdx.x * blockDim.x + threadIdx.x; if (i < n) p[i] = v;
}
__global__ void degree_kernel(const int* __restrict__ dst, int* __restrict__ cnt) {
  int e = blockIdx.x * blockDim.x + threadIdx.x;
  if (e < NE) atomicAdd(&cnt[dst[e]], 1);
}
// h[n,384] (+)= [sum | fixed-max | mean] per unshared block ; xr = relu(h)
__global__ void finalize_kernel(const float* __restrict__ Ssum, const float* __restrict__ Smax,
                                const int* __restrict__ cnt, float* __restrict__ h,
                                float* __restrict__ xr, int addTo) {
  int i = blockIdx.x * blockDim.x + threadIdx.x;
  if (i >= NN * 384) return;
  int n = i / 384, col = i % 384;
  int u = col / 96, k2 = col % 96, kind = k2 / 32, c = k2 & 31;
  float s = Ssum[(size_t)n * 128 + u * 32 + c];
  float v;
  if (kind == 0) v = s;
  else if (kind == 1) {
    float mm = Smax[(size_t)n * 128 + u * 32 + c];
    v = (mm >= -3.3e38f) ? mm : 0.f;   // empty segment (-inf) / non-finite -> 0
  } else {
    v = s / fmaxf((float)cnt[n], 1.f);
  }
  float hv = addTo ? (h[i] + v) : v;
  h[i] = hv;
  xr[i] = fmaxf(hv, 0.f);
}

// ---------------- host driver ------------------------------------------------
extern "C" void kernel_launch(void* const* d_in, const int* in_sizes, int n_in,
                              void* d_out, int out_size, void* d_ws, size_t ws_size,
                              hipStream_t stream) {
  (void)out_size; (void)ws_size;
  // Locate pytree leaves by flat size (robust to sorted vs insertion flatten).
  const float* x = nullptr; const int* ei = nullptr;
  int w1L0[4] = {0,0,0,0}, w1L1[4] = {0,0,0,0}; int c0 = 0, c1 = 0, iFC = -1, iGH = -1;
  for (int i = 0; i < n_in; ++i) {
    switch (in_sizes[i]) {
      case 140000: x = (const float*)d_in[i]; break;
      case 320000: ei = (const int*)d_in[i]; break;
      case 768:    if (c0 < 4) w1L0[c0++] = i; break;
      case 49024:  if (c1 < 4) w1L1[c1++] = i; break;
      case 12288:  iFC = i; break;
      case 224:    iGH = i; break;
      default: break;
    }
  }
  const int* srcI = ei;
  const int* dstI = ei + NE;

  auto getMlp = [&](int i0, int w2sz) -> MlpHost {
    MlpHost r;
    auto F = [&](int k) { return (const float*)d_in[k]; };
    if (i0 + 1 < n_in && in_sizes[i0 + 1] == w2sz) {
      // jax sorted-key flatten: W1,W2,W3,b1,b2,b3,be1,be2,g1,g2
      r.W1=F(i0);   r.W2=F(i0+1); r.W3=F(i0+2); r.b1=F(i0+3); r.b2=F(i0+4);
      r.b3=F(i0+5); r.be1=F(i0+6); r.be2=F(i0+7); r.g1=F(i0+8); r.g2=F(i0+9);
    } else {
      // insertion order: W1,b1,g1,be1,W2,b2,g2,be2,W3,b3
      r.W1=F(i0);   r.b1=F(i0+1); r.g1=F(i0+2); r.be1=F(i0+3); r.W2=F(i0+4);
      r.b2=F(i0+5); r.g2=F(i0+6); r.be2=F(i0+7); r.W3=F(i0+8); r.b3=F(i0+9);
    }
    return r;
  };

  // Workspace layout (bytes)
  char* ws = (char*)d_ws;
  float* XR  = (float*)(ws);                    // NN*384 f32
  float* H   = (float*)(ws + 30720000);         // NN*384 f32
  float* SS  = (float*)(ws + 61440000);         // NN*128 f32 (sum)
  float* SM  = (float*)(ws + 71680000);         // NN*128 f32 (max)
  int*   CNT = (int*)  (ws + 81920000);         // NN i32
  unsigned int* WP = (unsigned int*)(ws + 82000000);
  size_t wpo = 0;
  auto pack = [&](const float* W, int K, int N, int KT, int NT) -> const unsigned int* {
    unsigned int* dp = WP + wpo;
    int total = KT * NT * 256;
    wpo += (size_t)total;
    pack_w_kernel<<<(total + 255) / 256, 256, 0, stream>>>(W, dp, K, N, KT, NT);
    return dp;
  };
  auto fillMlp = [&](MlpPtrs& d, const MlpHost& m, int K1, int KT1w, int N3, int NT3) {
    d.w1 = pack(m.W1, K1, 64, KT1w, 4);
    d.w2 = pack(m.W2, 64, 64, 2, 4);
    d.w3 = pack(m.W3, 64, N3, 2, NT3);
    d.b1 = m.b1; d.g1 = m.g1; d.be1 = m.be1;
    d.b2 = m.b2; d.g2 = m.g2; d.be2 = m.be2; d.b3 = m.b3;
  };

  EdgeParams EP0, EP1; NodeParams NP;
  for (int u = 0; u < 4; ++u) {
    MlpHost m0 = getMlp(w1L0[u], 4096);
    fillMlp(EP0.m[u], m0, 12, 1, 32, 2);
    MlpHost m1 = getMlp(w1L1[u], 4096);
    fillMlp(EP1.m[u], m1, 766, 24, 32, 2);
  }
  {
    MlpHost f = getMlp(iFC, 1024);
    NP.fc.w1 = pack(f.W1, 384, 32, 12, 2);
    NP.fc.w2 = pack(f.W2, 32, 32, 1, 2);
    NP.fc.w3 = pack(f.W3, 32, 4, 1, 1);
    NP.fc.b1 = f.b1; NP.fc.g1 = f.g1; NP.fc.be1 = f.be1;
    NP.fc.b2 = f.b2; NP.fc.g2 = f.g2; NP.fc.be2 = f.be2; NP.fc.b3 = f.b3;
    MlpHost g = getMlp(iGH, 1024);
    NP.gh.w1 = pack(g.W1, 7, 32, 1, 2);
    NP.gh.w2 = pack(g.W2, 32, 32, 1, 2);
    NP.gh.w3 = pack(g.W3, 32, 4, 1, 1);
    NP.gh.b1 = g.b1; NP.gh.g1 = g.g1; NP.gh.be1 = g.be1;
    NP.gh.b2 = g.b2; NP.gh.g2 = g.g2; NP.gh.be2 = g.be2; NP.gh.b3 = g.b3;
  }

  const int SSN = NN * 128;
  const int HN  = NN * 384;
  fill_i32<<<(NN + 255) / 256, 256, 0, stream>>>(CNT, 0, NN);
  degree_kernel<<<(NE + 255) / 256, 256, 0, stream>>>(dstI, CNT);

  // ---- layer 0 ----
  fill_f32<<<(SSN + 255) / 256, 256, 0, stream>>>(SS, 0.f, SSN);
  fill_f32<<<(SSN + 255) / 256, 256, 0, stream>>>(SM, -INFINITY, SSN);
  edge_mlp_kernel<7, 1><<<NE / 32, 64, 0, stream>>>(x, srcI, dstI, EP0, SS, SM);
  finalize_kernel<<<(HN + 255) / 256, 256, 0, stream>>>(SS, SM, CNT, H, XR, 0);

  // ---- layer 1 (residual) ----
  fill_f32<<<(SSN + 255) / 256, 256, 0, stream>>>(SS, 0.f, SSN);
  fill_f32<<<(SSN + 255) / 256, 256, 0, stream>>>(SM, -INFINITY, SSN);
  edge_mlp_kernel<384, 24><<<NE / 32, 64, 0, stream>>>(XR, srcI, dstI, EP1, SS, SM);
  finalize_kernel<<<(HN + 255) / 256, 256, 0, stream>>>(SS, SM, CNT, H, XR, 1);

  // ---- output heads ----
  node_out_kernel<<<NN / 32, 64, 0, stream>>>(XR, x, NP, (float*)d_out);
}
